// MulLSTM_64304250355912
// MI455X (gfx1250) — matrix operand order repaired and checked by reading, not compiled, not run on hardware
//
#include <hip/hip_runtime.h>
#include <hip/hip_bf16.h>

typedef __bf16 bf16;
typedef __attribute__((ext_vector_type(16))) __bf16 v16bf;
typedef __attribute__((ext_vector_type(8)))  __bf16 v8bf;
typedef __attribute__((ext_vector_type(8)))  float  v8f;

#define BATCH 64
#define SEQ   512
#define IN    512
#define HID   1024
#define CLS   128
#define KCAT  (IN + HID)     // 1536
#define G4    (4 * HID)      // 4096
#define XSTEP (16 * 4 * 32 * 16)   // packed A elems per step slice of x (64x512 bf16)

union FragBF {
    v16bf v;
    struct { v8bf lo; v8bf hi; } s;
};

__device__ __forceinline__ float sigf(float x) { return 1.0f / (1.0f + __expf(-x)); }

// ---------------------------------------------------------------------------
// Fragment layouts (wave32 WMMA 16x16x32 bf16, this code's contiguous k order):
//  A frag slot: idx(kt,mt,lane,j) = ((kt*4 + mt)*32 + lane)*16 + j
//     lane = hf*16 + (m&15), k = kt*32 + hf*16 + j
//  B frag slot: idx(kt,nt,lane,j) = ((kt*NT + nt)*32 + lane)*16 + j
//     lane = hf*16 + (n&15), k = kt*32 + hf*16 + j
// ---------------------------------------------------------------------------
__device__ __forceinline__ size_t afrag_idx(int m, int kk, int kt) {
    int mt = m >> 4, mr = m & 15;
    int hf = (kk >> 4) & 1;
    int j  = kk & 15;
    int lane = hf * 16 + mr;
    return ((size_t)(kt * 4 + mt) * 32 + lane) * 16 + j;
}

// ------------------------------- weight packs ------------------------------
__global__ void pack_wm(const float* __restrict__ Wm, const float* __restrict__ Um,
                        bf16* __restrict__ dst) {
    const int NT = IN / 16;
    const int total = (KCAT / 32) * NT * 32;
    for (int s = blockIdx.x * blockDim.x + threadIdx.x; s < total;
         s += gridDim.x * blockDim.x) {
        int lane = s & 31, f = s >> 5;
        int kt = f / NT, nt = f % NT;
        int n  = nt * 16 + (lane & 15);
        int kb = kt * 32 + (lane >> 4) * 16;
        bf16* out = dst + (size_t)s * 16;
        for (int j = 0; j < 16; ++j) {
            int k = kb + j;
            float v = (k < IN) ? Wm[(size_t)k * IN + n] : Um[(size_t)(k - IN) * IN + n];
            out[j] = (bf16)v;
        }
    }
}

__global__ void pack_wg(const float* __restrict__ Wi, const float* __restrict__ Wf,
                        const float* __restrict__ Wo, const float* __restrict__ Wc,
                        const float* __restrict__ Ui, const float* __restrict__ Uf,
                        const float* __restrict__ Uo, const float* __restrict__ Uc,
                        bf16* __restrict__ dst) {
    const int NT = G4 / 16;
    const int total = (KCAT / 32) * NT * 32;
    for (int s = blockIdx.x * blockDim.x + threadIdx.x; s < total;
         s += gridDim.x * blockDim.x) {
        int lane = s & 31, f = s >> 5;
        int kt = f / NT, nt = f % NT;
        int n  = nt * 16 + (lane & 15);
        int gate = n >> 10, col = n & 1023;
        const float* W = gate == 0 ? Wi : gate == 1 ? Wf : gate == 2 ? Wo : Wc;
        const float* U = gate == 0 ? Ui : gate == 1 ? Uf : gate == 2 ? Uo : Uc;
        int kb = kt * 32 + (lane >> 4) * 16;
        bf16* out = dst + (size_t)s * 16;
        for (int j = 0; j < 16; ++j) {
            int k = kb + j;
            float v = (k < IN) ? W[(size_t)k * HID + col] : U[(size_t)(k - IN) * HID + col];
            out[j] = (bf16)v;
        }
    }
}

__global__ void pack_lin(const float* __restrict__ linW, bf16* __restrict__ dst) {
    const int NT = CLS / 16;
    const int total = (HID / 32) * NT * 32;
    for (int s = blockIdx.x * blockDim.x + threadIdx.x; s < total;
         s += gridDim.x * blockDim.x) {
        int lane = s & 31, f = s >> 5;
        int kt = f / NT, nt = f % NT;
        int n  = nt * 16 + (lane & 15);
        int kb = kt * 32 + (lane >> 4) * 16;
        bf16* out = dst + (size_t)s * 16;
        for (int j = 0; j < 16; ++j)
            out[j] = (bf16)linW[(size_t)n * HID + (kb + j)];   // B = lin_W^T
    }
}

// x (B,S,I) fp32 -> per-step A-fragment bf16 (once per call)
__global__ void pack_x(const float* __restrict__ x, bf16* __restrict__ xpack) {
    const size_t total = (size_t)BATCH * SEQ * IN;
    for (size_t idx = (size_t)blockIdx.x * blockDim.x + threadIdx.x; idx < total;
         idx += (size_t)gridDim.x * blockDim.x) {
        int i = (int)(idx % IN);
        size_t rest = idx / IN;
        int t = (int)(rest % SEQ);
        int b = (int)(rest / SEQ);
        xpack[(size_t)t * XSTEP + afrag_idx(b, i & 31, i >> 5)] = (bf16)x[idx];
    }
}

__global__ void init_state(bf16* hfrag, float* cst) {
    int i = blockIdx.x * blockDim.x + threadIdx.x;
    if (i < BATCH * HID) { hfrag[i] = (bf16)0.0f; cst[i] = 0.0f; }
}

// ---------------------------------------------------------------------------
// GEMM core segment: bases are per-lane pointers; every other address is a
// compile-time constant element offset (folds into load immediate offsets).
// Per kt: 4 A b128 + 8 B b128 loads, 8 WMMAs (wave tile 32x64).
// ---------------------------------------------------------------------------
template <int NKT, int NT>
__device__ __forceinline__ void gemm_seg(const bf16* __restrict__ abase,
                                         const bf16* __restrict__ bbase,
                                         v8f acc[2][4]) {
    __builtin_prefetch(bbase, 0, 3);
#pragma unroll
    for (int kt = 0; kt < NKT; ++kt) {
        FragBF a[2], b[4];
#pragma unroll
        for (int mi = 0; mi < 2; ++mi) {
            const v8bf* ap = (const v8bf*)(abase + kt * 2048 + mi * 512);
            a[mi].s.lo = ap[0];
            a[mi].s.hi = ap[1];
        }
#pragma unroll
        for (int ni = 0; ni < 4; ++ni) {
            const v8bf* bp = (const v8bf*)(bbase + kt * (NT * 512) + ni * 512);
            b[ni].s.lo = bp[0];
            b[ni].s.hi = bp[1];
        }
#pragma unroll
        for (int mi = 0; mi < 2; ++mi)
#pragma unroll
            for (int ni = 0; ni < 4; ++ni)
                acc[mi][ni] = __builtin_amdgcn_wmma_f32_16x16x32_bf16(
                    false, a[mi].v, false, b[ni].v, (short)0, acc[mi][ni], false, false);
    }
}

// common per-kernel lane setup: 128 threads = 4 waves; wave tile 32x64
#define GEMM_SETUP(NTOT)                                                     \
    const int tid = threadIdx.x;                                             \
    const unsigned lane = tid & 31, wave = tid >> 5;                         \
    const unsigned mh = wave & 1, nq = wave >> 1;                            \
    const unsigned nt0 = blockIdx.x * 8 + nq * 4;                            \
    const int hf = lane >> 4;                                                \
    (void)hf;                                                                \
    v8f acc[2][4];                                                           \
    _Pragma("unroll") for (int mi = 0; mi < 2; ++mi)                         \
        _Pragma("unroll") for (int ni = 0; ni < 4; ++ni)                     \
            _Pragma("unroll") for (int r = 0; r < 8; ++r) acc[mi][ni][r] = 0.0f;

// GEMM1: m = [x_t | h] @ [W_m;U_m] + b_m ; xm = x_t*m -> xmfrag.
// M=64, N=512, K=1536. grid=4 x 128.
__global__ void __launch_bounds__(128)
gemm_m(const bf16* __restrict__ xpack, const bf16* __restrict__ hfrag,
       const bf16* __restrict__ wmp, const float* __restrict__ bm,
       bf16* __restrict__ xmfrag, int step) {
    GEMM_SETUP(IN / 16);
    const bf16* xs = xpack + (size_t)step * XSTEP;
    const bf16* a0 = xs + ((size_t)(mh * 2) * 32 + lane) * 16;
    const bf16* a1 = hfrag + ((size_t)(mh * 2) * 32 + lane) * 16;
    const bf16* b0 = wmp + ((size_t)nt0 * 32 + lane) * 16;
    const bf16* b1 = wmp + ((size_t)(16 * (IN / 16) + nt0) * 32 + lane) * 16;
    gemm_seg<16, IN / 16>(a0, b0, acc);
    gemm_seg<32, IN / 16>(a1, b1, acc);

#pragma unroll
    for (int ni = 0; ni < 4; ++ni) {
        const unsigned n = (nt0 + ni) * 16 + (lane & 15);
        const float bmv = bm[n];
        const int ktx = n >> 5, kk = n & 31;
        const int hf2 = (kk >> 4) & 1;
        const int j   = kk & 15;
        // element (m, n): m = (mh*2+mi)*16 + hf*8 + r
        bf16* st = xmfrag + ((size_t)(ktx * 4 + mh * 2) * 32 + hf2 * 16 + hf * 8) * 16 + j;
        const bf16* xl = xs + ((size_t)(ktx * 4 + mh * 2) * 32 + hf2 * 16 + hf * 8) * 16 + j;
#pragma unroll
        for (int mi = 0; mi < 2; ++mi)
#pragma unroll
            for (int r = 0; r < 8; ++r) {
                float mval = acc[mi][ni][r] + bmv;
                float xv = (float)xl[mi * 512 + r * 16];
                st[mi * 512 + r * 16] = (bf16)(xv * mval);
            }
    }
}

// GEMM2: g = [xm | h] @ [Wg;Ug] + bg. M=64, N=4096, K=1536. grid=32 x 128.
__global__ void __launch_bounds__(128)
gemm_g(const bf16* __restrict__ xmfrag, const bf16* __restrict__ hfrag,
       const bf16* __restrict__ wgp,
       const float* __restrict__ bi, const float* __restrict__ bfb,
       const float* __restrict__ bo, const float* __restrict__ bc,
       float* __restrict__ gbuf) {
    GEMM_SETUP(G4 / 16);
    const bf16* a0 = xmfrag + ((size_t)(mh * 2) * 32 + lane) * 16;
    const bf16* a1 = hfrag + ((size_t)(mh * 2) * 32 + lane) * 16;
    const bf16* b0 = wgp + ((size_t)nt0 * 32 + lane) * 16;
    const bf16* b1 = wgp + ((size_t)(16 * (G4 / 16) + nt0) * 32 + lane) * 16;
    gemm_seg<16, G4 / 16>(a0, b0, acc);
    gemm_seg<32, G4 / 16>(a1, b1, acc);

#pragma unroll
    for (int ni = 0; ni < 4; ++ni) {
        const unsigned n = (nt0 + ni) * 16 + (lane & 15);
        const int gate = n >> 10, col = n & 1023;
        const float bias = (gate == 0 ? bi : gate == 1 ? bfb : gate == 2 ? bo : bc)[col];
        float* st = gbuf + (size_t)(mh * 32 + hf * 8) * G4 + n;
#pragma unroll
        for (int mi = 0; mi < 2; ++mi)
#pragma unroll
            for (int r = 0; r < 8; ++r)
                st[(size_t)(mi * 16 + r) * G4] = acc[mi][ni][r] + bias;
    }
}

// gates -> c,h ; h scattered into fragment layout
__global__ void lstm_update(const float* __restrict__ gbuf, float* __restrict__ cst,
                            bf16* __restrict__ hfrag) {
    int i = blockIdx.x * blockDim.x + threadIdx.x;
    if (i >= BATCH * HID) return;
    int b = i >> 10, j = i & 1023;
    const float* g = gbuf + (size_t)b * G4;
    float ig = g[j], fg = g[j + HID], og = g[j + 2 * HID], cgt = g[j + 3 * HID];
    float c = sigf(fg) * cst[i] + sigf(ig) * tanhf(cgt);
    float h = sigf(og) * tanhf(c);
    cst[i] = c;
    hfrag[afrag_idx(b, j & 31, j >> 5)] = (bf16)h;
}

// out[:,t,:] = h @ lin_W^T + lin_b. M=64, N=128, K=1024. grid=1 x 128.
__global__ void __launch_bounds__(128)
gemm_out(const bf16* __restrict__ hfrag, const bf16* __restrict__ lwp,
         const float* __restrict__ lb, float* __restrict__ out, int step) {
    GEMM_SETUP(CLS / 16);
    const bf16* a0 = hfrag + ((size_t)(mh * 2) * 32 + lane) * 16;
    const bf16* b0 = lwp + ((size_t)nt0 * 32 + lane) * 16;
    gemm_seg<32, CLS / 16>(a0, b0, acc);

#pragma unroll
    for (int ni = 0; ni < 4; ++ni) {
        const unsigned n = (nt0 + ni) * 16 + (lane & 15);
        const float lbv = lb[n];
        float* st = out + ((size_t)(mh * 32 + hf * 8) * SEQ + step) * CLS + n;
#pragma unroll
        for (int mi = 0; mi < 2; ++mi)
#pragma unroll
            for (int r = 0; r < 8; ++r)
                st[(size_t)(mi * 16 + r) * SEQ * CLS] = acc[mi][ni][r] + lbv;
    }
}

// ---------------------------------------------------------------------------
extern "C" void kernel_launch(void* const* d_in, const int* in_sizes, int n_in,
                              void* d_out, int out_size, void* d_ws, size_t ws_size,
                              hipStream_t stream) {
    const float* x    = (const float*)d_in[0];
    const float* W_i  = (const float*)d_in[1];
    const float* W_f  = (const float*)d_in[2];
    const float* W_o  = (const float*)d_in[3];
    const float* W_c  = (const float*)d_in[4];
    const float* W_m  = (const float*)d_in[5];
    const float* U_i  = (const float*)d_in[6];
    const float* U_f  = (const float*)d_in[7];
    const float* U_o  = (const float*)d_in[8];
    const float* U_c  = (const float*)d_in[9];
    const float* U_m  = (const float*)d_in[10];
    const float* b_i  = (const float*)d_in[11];
    const float* b_f  = (const float*)d_in[12];
    const float* b_o  = (const float*)d_in[13];
    const float* b_c  = (const float*)d_in[14];
    const float* b_m  = (const float*)d_in[15];
    const float* linW = (const float*)d_in[16];
    const float* linb = (const float*)d_in[17];
    float* out = (float*)d_out;

    char* ws = (char*)d_ws;
    size_t off = 0;
    auto alloc = [&](size_t bytes) -> void* {
        void* p = ws + off;
        off += (bytes + 255) & ~(size_t)255;
        return p;
    };
    bf16*  wmp    = (bf16*)alloc((size_t)KCAT * IN * sizeof(bf16));
    bf16*  wgp    = (bf16*)alloc((size_t)KCAT * G4 * sizeof(bf16));
    bf16*  lwp    = (bf16*)alloc((size_t)HID * CLS * sizeof(bf16));
    bf16*  hfrag  = (bf16*)alloc((size_t)BATCH * HID * sizeof(bf16));
    bf16*  xmfrag = (bf16*)alloc((size_t)BATCH * IN * sizeof(bf16));
    float* cst    = (float*)alloc((size_t)BATCH * HID * sizeof(float));
    float* gbuf   = (float*)alloc((size_t)BATCH * G4 * sizeof(float));
    bf16*  xpack  = (bf16*)alloc((size_t)SEQ * XSTEP * sizeof(bf16));

    // one-time (per call) packing; weights + x become L2-resident bf16 fragments
    pack_wm<<<192, 256, 0, stream>>>(W_m, U_m, wmp);
    pack_wg<<<1024, 256, 0, stream>>>(W_i, W_f, W_o, W_c, U_i, U_f, U_o, U_c, wgp);
    pack_lin<<<32, 256, 0, stream>>>(linW, lwp);
    pack_x<<<4096, 256, 0, stream>>>(x, xpack);
    init_state<<<(BATCH * HID + 255) / 256, 256, 0, stream>>>(hfrag, cst);

    for (int t = 0; t < SEQ; ++t) {
        gemm_m<<<4, 128, 0, stream>>>(xpack, hfrag, wmp, b_m, xmfrag, t);
        gemm_g<<<32, 128, 0, stream>>>(xmfrag, hfrag, wgp, b_i, b_f, b_o, b_c, gbuf);
        lstm_update<<<256, 256, 0, stream>>>(gbuf, cst, hfrag);
        gemm_out<<<1, 128, 0, stream>>>(hfrag, lwp, linb, out, t);
    }
}
